// SceneEncoder_47708496724719
// MI455X (gfx1250) — matrix operand order repaired and checked by reading, not compiled, not run on hardware
//
#include <hip/hip_runtime.h>
#include <hip/hip_bf16.h>
#include <math.h>

#define HH 8
#define CC 16
#define HID 128
#define IN0 288
#define RELN 51
#define NEG_SLOPE 0.2f
#define BN_EPS 1e-5f

typedef __bf16 bf16x16 __attribute__((ext_vector_type(16)));
typedef float  f32x8   __attribute__((ext_vector_type(8)));
typedef unsigned short u16x8 __attribute__((ext_vector_type(8)));

union V16 {
  bf16x16 bf;
  u16x8   h[2];
};

__device__ __forceinline__ unsigned short f2bf(float f) {
  unsigned u = __float_as_uint(f);
  unsigned r = 0x7FFFu + ((u >> 16) & 1u);
  return (unsigned short)((u + r) >> 16);
}

// float atomic max via sign-split int/uint atomics (init value must be -inf bits)
__device__ __forceinline__ void atomicMaxF(float* addr, float v) {
  if (v >= 0.0f) atomicMax((int*)addr, __float_as_int(v));
  else           atomicMin((unsigned int*)addr, __float_as_uint(v));
}

// ---------------------------------------------------------------------------
// Input feature build: x[N,288] bf16 = concat(tok_emb[tok_id], bbox@W+b)
// ---------------------------------------------------------------------------
__global__ void build_x(const int* __restrict__ tok_id, const float* __restrict__ bbox,
                        const float* __restrict__ tok_emb, const float* __restrict__ bbox_W,
                        const float* __restrict__ bbox_b, unsigned short* __restrict__ xbf,
                        int N) {
  int idx = blockIdx.x * blockDim.x + threadIdx.x;
  if (idx >= N * IN0) return;
  int n = idx / IN0, c = idx - n * IN0;
  float v;
  if (c < 256) {
    v = tok_emb[(long)tok_id[n] * 256 + c];
  } else {
    int j = c - 256;
    v = bbox_b[j];
#pragma unroll
    for (int k = 0; k < 4; ++k) v += bbox[(long)n * 4 + k] * bbox_W[k * 32 + j];
  }
  xbf[idx] = f2bf(v);
}

// relW[layer][r,:] = rel_emb[r,:] @ We_layer   -> [3,51,128]
__global__ void build_relw(const float* __restrict__ rel_emb, const float* __restrict__ We0,
                           const float* __restrict__ WeS, float* __restrict__ relW) {
  int idx = blockIdx.x * blockDim.x + threadIdx.x;
  if (idx >= 3 * RELN * HID) return;
  int l = idx / (RELN * HID);
  int rem = idx - l * RELN * HID;
  int r = rem / HID, o = rem - r * HID;
  const float* We = (l == 0) ? We0 : (WeS + (long)(l - 1) * CC * HID);
  float v = 0.f;
#pragma unroll
  for (int c = 0; c < CC; ++c) v += rel_emb[r * CC + c] * We[c * HID + o];
  relW[idx] = v;
}

// ---------------------------------------------------------------------------
// Pack W[K,128] f32 row-major into per-lane WMMA B layout (bf16):
// group (kt,nt): 32 lanes x 16 values; lane half selects K base (0/16), j contiguous.
// ---------------------------------------------------------------------------
__global__ void pack_w(const float* __restrict__ W, unsigned short* __restrict__ Wp, int K) {
  int idx = blockIdx.x * blockDim.x + threadIdx.x;
  if (idx >= K * HID) return;
  int j = idx & 15;
  int lane = (idx >> 4) & 31;
  int t = idx >> 9;            // kt*8 + nt
  int nt = t & 7, kt = t >> 3;
  int n = nt * 16 + (lane & 15);
  int k = kt * 32 + ((lane >> 4) << 4) + j;
  Wp[idx] = f2bf(W[(long)k * HID + n]);
}

// ---------------------------------------------------------------------------
// GEMM: out[M,128] = A[M,K](bf16) @ W(packed bf16) + bias,  f32 accumulate.
// One wave per 16x128 strip: A loaded once per K-step, 8 independent WMMAs
// (8 accumulators) -> high matrix-op density, 8x less A traffic.
// ---------------------------------------------------------------------------
__global__ void gemm_bf16_wmma(const unsigned short* __restrict__ A,
                               const unsigned short* __restrict__ Wp,
                               const float* __restrict__ bias,
                               float* __restrict__ out, int M, int K) {
  int wave = (int)((blockIdx.x * blockDim.x + threadIdx.x) >> 5);  // one wave per 16-row strip
  int lane = threadIdx.x & 31;
  int Mtiles = (M + 15) >> 4;
  if (wave >= Mtiles) return;   // wave-uniform exit: EXEC stays all-ones for WMMA

  int row = wave * 16 + (lane & 15);
  if (row >= M) row = M - 1;    // tail clamp (stores guarded below)
  const unsigned short* arow = A + (long)row * K + ((lane >> 4) << 4);

  f32x8 c0 = {}, c1 = {}, c2 = {}, c3 = {}, c4 = {}, c5 = {}, c6 = {}, c7 = {};
  int nkt = K >> 5;
  for (int kt = 0; kt < nkt; ++kt) {
    V16 a;
    a.h[0] = *(const u16x8*)(arow + kt * 32);
    a.h[1] = *(const u16x8*)(arow + kt * 32 + 8);
    const unsigned short* bp = Wp + (long)kt * 4096 + lane * 16;   // (kt*8+nt)*512 + lane*16
    V16 b;
#define GEMM_STEP(i, ci)                                                        \
    b.h[0] = *(const u16x8*)(bp + (i) * 512);                                   \
    b.h[1] = *(const u16x8*)(bp + (i) * 512 + 8);                               \
    ci = __builtin_amdgcn_wmma_f32_16x16x32_bf16(false, a.bf, false, b.bf,      \
                                                 (short)0, ci, false, false);
    GEMM_STEP(0, c0) GEMM_STEP(1, c1) GEMM_STEP(2, c2) GEMM_STEP(3, c3)
    GEMM_STEP(4, c4) GEMM_STEP(5, c5) GEMM_STEP(6, c6) GEMM_STEP(7, c7)
#undef GEMM_STEP
  }

  int rbase = wave * 16 + ((lane >> 4) << 3);   // D layout: half-wave rows 0-7 / 8-15
  int coll = lane & 15;
#define GEMM_STORE(i, ci) {                                                     \
    int col = (i) * 16 + coll;                                                  \
    float bv = bias[col];                                                       \
    _Pragma("unroll")                                                           \
    for (int r = 0; r < 8; ++r) {                                               \
      int rr = rbase + r;                                                       \
      if (rr < M) out[(long)rr * HID + col] = ci[r] + bv;                       \
    } }
  GEMM_STORE(0, c0) GEMM_STORE(1, c1) GEMM_STORE(2, c2) GEMM_STORE(3, c3)
  GEMM_STORE(4, c4) GEMM_STORE(5, c5) GEMM_STORE(6, c6) GEMM_STORE(7, c7)
#undef GEMM_STORE
}

__global__ void fill_f32(float* __restrict__ p, float v, int n) {
  int i = blockIdx.x * blockDim.x + threadIdx.x;
  if (i < n) p[i] = v;
}

// ---------------------------------------------------------------------------
// Edge pass 1: logits[e,h] = sum_c leaky(xl[src]+xr[dst]+relW[ea]) * att[h,c]
//              amax[dst,h] = max(...)
// ---------------------------------------------------------------------------
__global__ void edge_logits(const int* __restrict__ src, const int* __restrict__ dst,
                            const int* __restrict__ ea,
                            const float* __restrict__ xl, const float* __restrict__ xr,
                            const float* __restrict__ relW, const float* __restrict__ att,
                            float* __restrict__ logits, float* __restrict__ amax, int E) {
  int t = blockIdx.x * blockDim.x + threadIdx.x;
  if (t >= E * HH) return;
  int e = t >> 3, h = t & 7;
  int s = src[e], d = dst[e], r = ea[e];
  const float4* pl = (const float4*)(xl + (long)s * HID + h * CC);
  const float4* pr = (const float4*)(xr + (long)d * HID + h * CC);
  const float4* pe = (const float4*)(relW + (long)r * HID + h * CC);
  const float4* pa = (const float4*)(att + h * CC);
  float acc = 0.f;
#pragma unroll
  for (int q = 0; q < 4; ++q) {
    float4 a = pl[q], b = pr[q], ce = pe[q], w = pa[q];
    float m;
    m = a.x + b.x + ce.x; m = m > 0.f ? m : m * NEG_SLOPE; acc += m * w.x;
    m = a.y + b.y + ce.y; m = m > 0.f ? m : m * NEG_SLOPE; acc += m * w.y;
    m = a.z + b.z + ce.z; m = m > 0.f ? m : m * NEG_SLOPE; acc += m * w.z;
    m = a.w + b.w + ce.w; m = m > 0.f ? m : m * NEG_SLOPE; acc += m * w.w;
  }
  logits[t] = acc;
  atomicMaxF(amax + (long)d * HH + h, acc);
}

// Edge pass 2: p = exp(logit - amax[dst]); denom[dst] += p (p overwrites logits)
__global__ void edge_softmax_num(const int* __restrict__ dst, float* __restrict__ lp,
                                 const float* __restrict__ amax, float* __restrict__ denom,
                                 int E) {
  int t = blockIdx.x * blockDim.x + threadIdx.x;
  if (t >= E * HH) return;
  int e = t >> 3, h = t & 7;
  int d = dst[e];
  float p = __expf(lp[t] - amax[(long)d * HH + h]);
  lp[t] = p;
  atomicAdd(denom + (long)d * HH + h, p);
}

// Edge pass 3: acc[dst] += (p/denom[dst]) * xl[src]
__global__ void edge_scatter(const int* __restrict__ src, const int* __restrict__ dst,
                             const float* __restrict__ lp, const float* __restrict__ denom,
                             const float* __restrict__ xl, float* __restrict__ acc, int E) {
  int t = blockIdx.x * blockDim.x + threadIdx.x;
  if (t >= E * HH) return;
  int e = t >> 3, h = t & 7;
  int s = src[e], d = dst[e];
  float dn = denom[(long)d * HH + h];
  dn = dn > 1e-16f ? dn : 1e-16f;
  float alpha = lp[t] / dn;
  const float4* pl = (const float4*)(xl + (long)s * HID + h * CC);
  float* pa = acc + (long)d * HID + h * CC;
#pragma unroll
  for (int q = 0; q < 4; ++q) {
    float4 v = pl[q];
    atomicAdd(pa + q * 4 + 0, alpha * v.x);
    atomicAdd(pa + q * 4 + 1, alpha * v.y);
    atomicAdd(pa + q * 4 + 2, alpha * v.z);
    atomicAdd(pa + q * 4 + 3, alpha * v.w);
  }
}

// bias + relu (in place) and per-channel sum/sumsq -> stats[0:128]=sum, [128:256]=sumsq
__global__ void node_bias_relu_stats(float* __restrict__ acc, const float* __restrict__ bias,
                                     float* __restrict__ stats, int N) {
  int c = threadIdx.x;                 // 128 threads
  int r0 = blockIdx.x * 64;
  if (r0 >= N) return;
  int r1 = r0 + 64; if (r1 > N) r1 = N;
  float bv = bias[c];
  float s = 0.f, s2 = 0.f;
  for (int r = r0; r < r1; ++r) {
    float v = acc[(long)r * HID + c] + bv;
    v = v > 0.f ? v : 0.f;
    acc[(long)r * HID + c] = v;
    s += v; s2 += v * v;
  }
  atomicAdd(stats + c, s);
  atomicAdd(stats + HID + c, s2);
}

// stats -> scale/shift (biased variance, matching jnp.var)
__global__ void bn_finalize(float* __restrict__ stats, const float* __restrict__ gamma,
                            const float* __restrict__ beta, int N) {
  int c = threadIdx.x;
  float mu = stats[c] / (float)N;
  float var = stats[HID + c] / (float)N - mu * mu;
  float sc = gamma[c] * rsqrtf(var + BN_EPS);
  stats[c] = sc;
  stats[HID + c] = beta[c] - mu * sc;
}

// h = (residual ? h : 0) + scale*t + shift ; also emit bf16 copy for next GEMM
__global__ void bn_apply(const float* __restrict__ t, const float* __restrict__ stats,
                         float* __restrict__ h, unsigned short* __restrict__ hbf,
                         int N, int residual) {
  int idx = blockIdx.x * blockDim.x + threadIdx.x;
  if (idx >= N * HID) return;
  int c = idx & (HID - 1);
  float v = stats[c] * t[idx] + stats[HID + c];
  if (residual) v += h[idx];
  h[idx] = v;
  hbf[idx] = f2bf(v);
}

// segment mean-pool over sorted batch ids (run-length local accumulation + atomic flush)
__global__ void pool_sum(const float* __restrict__ h, const int* __restrict__ batch,
                         float* __restrict__ sums, float* __restrict__ counts, int N) {
  int c = threadIdx.x;                 // 128 threads
  int r0 = blockIdx.x * 64;
  if (r0 >= N) return;
  int r1 = r0 + 64; if (r1 > N) r1 = N;
  int curg = batch[r0];
  float s = 0.f, cnt = 0.f;
  for (int r = r0; r < r1; ++r) {
    int g = batch[r];
    if (g != curg) {
      atomicAdd(sums + (long)curg * HID + c, s);
      if (c == 0) atomicAdd(counts + curg, cnt);
      s = 0.f; cnt = 0.f; curg = g;
    }
    s += h[(long)r * HID + c];
    cnt += 1.f;
  }
  atomicAdd(sums + (long)curg * HID + c, s);
  if (c == 0) atomicAdd(counts + curg, cnt);
}

__global__ void pool_final(const float* __restrict__ sums, const float* __restrict__ counts,
                           float* __restrict__ out, int G) {
  int idx = blockIdx.x * blockDim.x + threadIdx.x;
  if (idx >= G * HID) return;
  float cn = counts[idx >> 7];
  cn = cn > 1.f ? cn : 1.f;
  out[idx] = sums[idx] / cn;
}

// ---------------------------------------------------------------------------
extern "C" void kernel_launch(void* const* d_in, const int* in_sizes, int n_in,
                              void* d_out, int out_size, void* d_ws, size_t ws_size,
                              hipStream_t stream) {
  (void)n_in; (void)ws_size;
  const int*   tok_id  = (const int*)d_in[0];
  const float* bbox    = (const float*)d_in[1];
  const int*   eidx    = (const int*)d_in[2];
  const int*   eattr   = (const int*)d_in[3];
  const int*   batch   = (const int*)d_in[4];
  const float* tok_emb = (const float*)d_in[5];
  const float* bbox_W  = (const float*)d_in[6];
  const float* bbox_b  = (const float*)d_in[7];
  const float* rel_emb = (const float*)d_in[8];
  const float* Wl0   = (const float*)d_in[9];
  const float* bl0   = (const float*)d_in[10];
  const float* Wr0   = (const float*)d_in[11];
  const float* br0   = (const float*)d_in[12];
  const float* We0   = (const float*)d_in[13];
  const float* att0  = (const float*)d_in[14];
  const float* bias0 = (const float*)d_in[15];
  const float* gamma0= (const float*)d_in[16];
  const float* beta0 = (const float*)d_in[17];
  const float* WlS   = (const float*)d_in[18];
  const float* blS   = (const float*)d_in[19];
  const float* WrS   = (const float*)d_in[20];
  const float* brS   = (const float*)d_in[21];
  const float* WeS   = (const float*)d_in[22];
  const float* attS  = (const float*)d_in[23];
  const float* biasS = (const float*)d_in[24];
  const float* gammaS= (const float*)d_in[25];
  const float* betaS = (const float*)d_in[26];

  const int N = in_sizes[0];
  const int E = in_sizes[3];
  const int G = out_size / HID;
  const int* src = eidx;
  const int* dst = eidx + E;

  // -------- workspace carve --------
  char* base = (char*)d_ws;
  size_t off = 0;
  auto carve = [&](size_t bytes) -> void* {
    void* p = (void*)(base + off);
    off += (bytes + 255) & ~(size_t)255;
    return p;
  };
  unsigned short* xbf  = (unsigned short*)carve((size_t)N * IN0 * 2);
  unsigned short* hbf  = (unsigned short*)carve((size_t)N * HID * 2);
  float* xl    = (float*)carve((size_t)N * HID * 4);
  float* xr    = (float*)carve((size_t)N * HID * 4);
  float* acc   = xr;  // alias: xr is dead after edge_logits, before acc is zeroed
  float* hbuf  = (float*)carve((size_t)N * HID * 4);
  float* ep    = (float*)carve((size_t)E * HH * 4);
  float* amax  = (float*)carve((size_t)N * HH * 4);
  float* denom = (float*)carve((size_t)N * HH * 4);
  float* relW  = (float*)carve((size_t)3 * RELN * HID * 4);
  float* stats = (float*)carve(256 * 4);
  unsigned short* Wp0l = (unsigned short*)carve((size_t)IN0 * HID * 2);
  unsigned short* Wp0r = (unsigned short*)carve((size_t)IN0 * HID * 2);
  unsigned short* WpSl = (unsigned short*)carve((size_t)2 * HID * HID * 2);
  unsigned short* WpSr = (unsigned short*)carve((size_t)2 * HID * HID * 2);
  float* sums   = (float*)carve((size_t)G * HID * 4);
  float* counts = (float*)carve((size_t)G * 4);

  // -------- preprocessing --------
  build_x<<<(N * IN0 + 255) / 256, 256, 0, stream>>>(tok_id, bbox, tok_emb, bbox_W, bbox_b, xbf, N);
  build_relw<<<(3 * RELN * HID + 255) / 256, 256, 0, stream>>>(rel_emb, We0, WeS, relW);
  pack_w<<<(IN0 * HID + 255) / 256, 256, 0, stream>>>(Wl0, Wp0l, IN0);
  pack_w<<<(IN0 * HID + 255) / 256, 256, 0, stream>>>(Wr0, Wp0r, IN0);
  for (int i = 0; i < 2; ++i) {
    pack_w<<<(HID * HID + 255) / 256, 256, 0, stream>>>(WlS + (long)i * HID * HID, WpSl + (long)i * HID * HID, HID);
    pack_w<<<(HID * HID + 255) / 256, 256, 0, stream>>>(WrS + (long)i * HID * HID, WpSr + (long)i * HID * HID, HID);
  }

  const int Mtiles = (N + 15) / 16;
  const int gemmBlocks = (Mtiles * 32 + 255) / 256;   // one wave per 16-row strip
  const int edgeBlocks = (E * HH + 255) / 256;
  const int rowBlocks  = (N + 63) / 64;

  // -------- 3 GATv2 layers --------
  for (int layer = 0; layer < 3; ++layer) {
    const unsigned short* inbf = (layer == 0) ? xbf : hbf;
    const int K = (layer == 0) ? IN0 : HID;
    const unsigned short* WpL = (layer == 0) ? Wp0l : WpSl + (long)(layer - 1) * HID * HID;
    const unsigned short* WpR = (layer == 0) ? Wp0r : WpSr + (long)(layer - 1) * HID * HID;
    const float* bl   = (layer == 0) ? bl0   : blS   + (long)(layer - 1) * HID;
    const float* br   = (layer == 0) ? br0   : brS   + (long)(layer - 1) * HID;
    const float* attp = (layer == 0) ? att0  : attS  + (long)(layer - 1) * HH * CC;
    const float* bp   = (layer == 0) ? bias0 : biasS + (long)(layer - 1) * HID;
    const float* gam  = (layer == 0) ? gamma0: gammaS+ (long)(layer - 1) * HID;
    const float* bet  = (layer == 0) ? beta0 : betaS + (long)(layer - 1) * HID;
    const float* relWl = relW + (long)layer * RELN * HID;

    gemm_bf16_wmma<<<gemmBlocks, 256, 0, stream>>>(inbf, WpL, bl, xl, N, K);
    gemm_bf16_wmma<<<gemmBlocks, 256, 0, stream>>>(inbf, WpR, br, xr, N, K);

    fill_f32<<<(N * HH + 255) / 256, 256, 0, stream>>>(amax, -INFINITY, N * HH);
    hipMemsetAsync(denom, 0, (size_t)N * HH * 4, stream);
    hipMemsetAsync(stats, 0, 256 * 4, stream);

    edge_logits<<<edgeBlocks, 256, 0, stream>>>(src, dst, eattr, xl, xr, relWl, attp, ep, amax, E);
    edge_softmax_num<<<edgeBlocks, 256, 0, stream>>>(dst, ep, amax, denom, E);
    hipMemsetAsync(acc, 0, (size_t)N * HID * 4, stream);   // xr dead from here on
    edge_scatter<<<edgeBlocks, 256, 0, stream>>>(src, dst, ep, denom, xl, acc, E);

    node_bias_relu_stats<<<rowBlocks, 128, 0, stream>>>(acc, bp, stats, N);
    bn_finalize<<<1, 128, 0, stream>>>(stats, gam, bet, N);
    bn_apply<<<(N * HID + 255) / 256, 256, 0, stream>>>(acc, stats, hbuf, hbf, N, layer > 0 ? 1 : 0);
  }

  // -------- global mean pool --------
  hipMemsetAsync(sums, 0, (size_t)G * HID * 4, stream);
  hipMemsetAsync(counts, 0, (size_t)G * 4, stream);
  pool_sum<<<rowBlocks, 128, 0, stream>>>(hbuf, batch, sums, counts, N);
  pool_final<<<(G * HID + 255) / 256, 256, 0, stream>>>(sums, counts, (float*)d_out, G);
}
